// VibeVoiceTextEncoder_52381421142357
// MI455X (gfx1250) — matrix operand order repaired and checked by reading, not compiled, not run on hardware
//
#include <hip/hip_runtime.h>
#include <math.h>

// ---------------------------------------------------------------------------
// VibeVoice text encoder for MI455X (gfx1250).
// bf16 WMMA (f32 accumulate) for all matmuls; weight tiles (64 cols x 64 K,
// 8 KB) double-buffered in LDS via global_load_async_to_lds_b128 with an
// s_wait_asynccnt + barrier handshake; 8 WMMAs per sync window.
// Workspace requirement ~700 MiB.
// ---------------------------------------------------------------------------

#define DEVI __device__ __forceinline__

typedef __attribute__((ext_vector_type(16))) __bf16 v16bf;
typedef __attribute__((ext_vector_type(8)))  __bf16 v8bf;
typedef __attribute__((ext_vector_type(8)))  float  v8f;

static constexpr int Bc = 4, Sc = 1024, Hc = 1024, NHc = 16, DHc = 64,
                     Ic = 4096, Lc = 4, Pc = 1024;

union FragU { v16bf v; v8bf h[2]; };

DEVI v8f zf8() {
  v8f z;
  #pragma unroll
  for (int i = 0; i < 8; ++i) z[i] = 0.0f;
  return z;
}

DEVI v16bf zb16() {
  v16bf z;
  #pragma unroll
  for (int i = 0; i < 16; ++i) z[i] = (__bf16)0.0f;
  return z;
}

// A-fragment (16x32 bf16): lanes 0-15 hold K {0..15}, lanes 16-31 hold
// K {16..31} of row m = lane&15.
DEVI v16bf load_a(const __bf16* __restrict__ A, int ld, int row, int kb, int half) {
  const __bf16* p = A + (size_t)row * ld + kb + half * 16;
  FragU u;
  u.h[0] = *(const v8bf*)p;
  u.h[1] = *(const v8bf*)(p + 8);
  return u.v;
}

// B-fragment (32x16) from row-major (N x K) weights in global memory.
DEVI v16bf load_b(const __bf16* __restrict__ W, int ld, int col, int kb, int half) {
  const __bf16* p = W + (size_t)col * ld + kb + half * 16;
  FragU u;
  u.h[0] = *(const v8bf*)p;
  u.h[1] = *(const v8bf*)(p + 8);
  return u.v;
}

// B-fragment from an LDS-staged 64x64 tile (row = weight column, 64 K elems).
DEVI v16bf load_b_lds(const __bf16* row_base, int ks, int half) {
  FragU u;
  u.h[0] = *(const v8bf*)(row_base + ks + 16 * half);
  u.h[1] = *(const v8bf*)(row_base + ks + 16 * half + 8);
  return u.v;
}

DEVI v8f wmma32(v16bf a, v16bf b, v8f c) {
  return __builtin_amdgcn_wmma_f32_16x16x32_bf16(
      false, a, false, b, (short)0, c, false, false);
}

// Async copy of 16 bytes global -> LDS; tracked with ASYNCcnt.
DEVI void async_g2l_b128(unsigned lds_addr, const void* gaddr) {
  asm volatile("global_load_async_to_lds_b128 %0, %1, off"
               :: "v"(lds_addr), "v"((unsigned long long)(uintptr_t)gaddr)
               : "memory");
}
DEVI void wait_async0() {
  asm volatile("s_wait_asynccnt 0" ::: "memory");
}

DEVI float wave_sum(float x) {
  #pragma unroll
  for (int o = 16; o > 0; o >>= 1) x += __shfl_xor(x, o, 32);
  return x;
}
DEVI float wave_max(float x) {
  #pragma unroll
  for (int o = 16; o > 0; o >>= 1) x = fmaxf(x, __shfl_xor(x, o, 32));
  return x;
}

// ---------------------------------------------------------------------------
// Elementwise helpers
// ---------------------------------------------------------------------------
__global__ void k_cast(const float* __restrict__ s, __bf16* __restrict__ d, long long n) {
  long long i = (long long)blockIdx.x * blockDim.x + threadIdx.x;
  if (i < n) d[i] = (__bf16)s[i];
}

// (L, Cout, Cin, 3) f32 -> (L, 3, Cout, Cin) bf16
__global__ void k_pack3(const float* __restrict__ s, __bf16* __restrict__ d,
                        int Cout, int Cin, long long n) {
  long long i = (long long)blockIdx.x * blockDim.x + threadIdx.x;
  if (i >= n) return;
  int t = (int)(i % 3);
  long long r = i / 3;
  int c = (int)(r % Cin);
  long long r2 = r / Cin;
  int o = (int)(r2 % Cout);
  int l = (int)(r2 / Cout);
  d[(((size_t)l * 3 + t) * Cout + o) * Cin + c] = (__bf16)s[i];
}

__global__ void k_embed(const int* __restrict__ ids, const float* __restrict__ emb,
                        float* __restrict__ x) {
  size_t i = (size_t)blockIdx.x * blockDim.x + threadIdx.x; // B*S*H total
  int t = (int)(i >> 10);
  int c = (int)(i & 1023);
  x[i] = emb[(size_t)ids[t] * Hc + c] * 32.0f; // sqrt(H)=32
}

// ---------------------------------------------------------------------------
// LayerNorm over H=1024, one block (256 thr) per token, output bf16
// ---------------------------------------------------------------------------
__global__ void k_layernorm(const float* __restrict__ x, const float* __restrict__ g,
                            const float* __restrict__ b, __bf16* __restrict__ out) {
  int t = blockIdx.x;
  int tid = threadIdx.x;
  const float4 v = ((const float4*)(x + (size_t)t * Hc))[tid];
  float s = v.x + v.y + v.z + v.w;
  float q = v.x * v.x + v.y * v.y + v.z * v.z + v.w * v.w;
  __shared__ float s1[8], s2[8];
  float ws = wave_sum(s), wq = wave_sum(q);
  if ((tid & 31) == 0) { s1[tid >> 5] = ws; s2[tid >> 5] = wq; }
  __syncthreads();
  float m = 0.f, qq = 0.f;
  #pragma unroll
  for (int i = 0; i < 8; ++i) { m += s1[i]; qq += s2[i]; }
  m *= (1.0f / Hc);
  float var = qq * (1.0f / Hc) - m * m;
  float r = rsqrtf(var + 1e-5f);
  int c = tid * 4;
  __bf16* o = out + (size_t)t * Hc + c;
  o[0] = (__bf16)((v.x - m) * r * g[c + 0] + b[c + 0]);
  o[1] = (__bf16)((v.y - m) * r * g[c + 1] + b[c + 1]);
  o[2] = (__bf16)((v.z - m) * r * g[c + 2] + b[c + 2]);
  o[3] = (__bf16)((v.w - m) * r * g[c + 3] + b[c + 3]);
}

// Stage one 64x64-K bf16 weight tile into LDS buffer (2 async b128 / thread).
// kb offset applied here (W is not pre-offset by kb); ld = K.
DEVI void stage64(__bf16 (*bsm)[64][64], int buf, const __bf16* W, int ld,
                  int nblk, int kb, int tid) {
  const int row = tid >> 2;
  const int off = (tid & 3) * 16;
  const __bf16* g = W + (size_t)(nblk + row) * ld + kb + off;
  async_g2l_b128((unsigned)(uintptr_t)&bsm[buf][row][off], g);
  async_g2l_b128((unsigned)(uintptr_t)&bsm[buf][row][off + 8], g + 8);
}

// ---------------------------------------------------------------------------
// Generic GEMM: C[M,N] = A[M,K] * W[N,K]^T   (A, W bf16 row-major)
// 8 waves/block (4x2); wave = 32x32 macro-tile. B tile (64 cols x 64 K, 8 KB)
// double-buffered in LDS via async global->LDS copies shared by 4 waves;
// 8 WMMAs per sync window. grid = (M/128, N/64).
// EPI: 0 bf16 row-major; 1 bf16 transposed to (B,H,S); 2 f32 resid += acc+bias.
// ---------------------------------------------------------------------------
template <int EPI>
__global__ void k_gemm(const __bf16* __restrict__ A, const __bf16* __restrict__ W,
                       const float* __restrict__ bias, __bf16* __restrict__ outb,
                       float* __restrict__ outf, int K, int N) {
  __shared__ __bf16 bsm[2][64][64]; // 16 KB double buffer
  const int tid = threadIdx.x, lane = tid & 31, wv = tid >> 5;
  const int half = lane >> 4, ln = lane & 15;
  const int m0 = blockIdx.x * 128 + (wv & 3) * 32;
  const int nblk = blockIdx.y * 64;
  const int n0 = nblk + (wv >> 2) * 32;

  stage64(bsm, 0, W, K, nblk, 0, tid);

  v8f acc[2][2];
  acc[0][0] = zf8(); acc[0][1] = zf8(); acc[1][0] = zf8(); acc[1][1] = zf8();
  int buf = 0;
  for (int kb = 0; kb < K; kb += 64) {
    wait_async0();
    __syncthreads();
    if (kb + 64 < K) stage64(bsm, buf ^ 1, W, K, nblk, kb + 64, tid);
    __builtin_prefetch(A + (size_t)(m0 + ln) * K + kb + 512, 0, 1);
    const int bcol = (wv >> 2) * 32 + ln;
    #pragma unroll
    for (int ks = 0; ks < 64; ks += 32) {
      v16bf a0 = load_a(A, K, m0 + ln, kb + ks, half);
      v16bf a1 = load_a(A, K, m0 + 16 + ln, kb + ks, half);
      v16bf b0 = load_b_lds(&bsm[buf][bcol][0], ks, half);
      v16bf b1 = load_b_lds(&bsm[buf][bcol + 16][0], ks, half);
      acc[0][0] = wmma32(a0, b0, acc[0][0]);
      acc[0][1] = wmma32(a0, b1, acc[0][1]);
      acc[1][0] = wmma32(a1, b0, acc[1][0]);
      acc[1][1] = wmma32(a1, b1, acc[1][1]);
    }
    __syncthreads();
    buf ^= 1;
  }
  #pragma unroll
  for (int mi = 0; mi < 2; ++mi)
    #pragma unroll
    for (int ni = 0; ni < 2; ++ni)
      #pragma unroll
      for (int r = 0; r < 8; ++r) {
        int row = m0 + mi * 16 + r + 8 * half;
        int col = n0 + ni * 16 + ln;
        float v = acc[mi][ni][r];
        if (EPI == 0) {
          outb[(size_t)row * N + col] = (__bf16)v;
        } else if (EPI == 1) {
          int bb = row >> 10, ss = row & 1023; // store vt[(b*H + col)*S + s]
          outb[((size_t)bb * Hc + col) * Sc + ss] = (__bf16)v;
        } else {
          size_t o = (size_t)row * N + col;
          outf[o] += v + bias[col];
        }
      }
}

// ---------------------------------------------------------------------------
// Attention scores: QK^T/8 + rel bias + mask.  grid = (S/128, S/64, B*NH)
// Rel trick: T = Qtile(16x64) x rel[r0..r0+31]^T (4 WMMAs), then
// scores_rel[i][j] = T[i][j-i+15] gathered with wave32 shuffles.
// ---------------------------------------------------------------------------
__global__ void k_scores(const __bf16* __restrict__ qm, const __bf16* __restrict__ km,
                         const __bf16* __restrict__ relw, const float* __restrict__ mask,
                         float* __restrict__ sc) {
  const int bh = blockIdx.z, b = bh >> 4, h = bh & 15;
  const __bf16* Q = qm + (size_t)b * Sc * Hc + h * DHc;
  const __bf16* Kp = km + (size_t)b * Sc * Hc + h * DHc;
  const int tid = threadIdx.x, lane = tid & 31, wv = tid >> 5;
  const int half = lane >> 4, ln = lane & 15;
  const int m0 = blockIdx.x * 128 + (wv & 3) * 32;
  const int n0 = blockIdx.y * 64 + (wv >> 2) * 32;
  v8f acc[2][2];
  acc[0][0] = zf8(); acc[0][1] = zf8(); acc[1][0] = zf8(); acc[1][1] = zf8();
  #pragma unroll
  for (int kb = 0; kb < DHc; kb += 32) {
    v16bf a0 = load_a(Q, Hc, m0 + ln, kb, half);
    v16bf a1 = load_a(Q, Hc, m0 + 16 + ln, kb, half);
    v16bf b0 = load_b(Kp, Hc, n0 + ln, kb, half);
    v16bf b1 = load_b(Kp, Hc, n0 + 16 + ln, kb, half);
    acc[0][0] = wmma32(a0, b0, acc[0][0]);
    acc[0][1] = wmma32(a0, b1, acc[0][1]);
    acc[1][0] = wmma32(a1, b0, acc[1][0]);
    acc[1][1] = wmma32(a1, b1, acc[1][1]);
  }
  #pragma unroll
  for (int mi = 0; mi < 2; ++mi) {
    #pragma unroll
    for (int ni = 0; ni < 2; ++ni) {
      const int q0 = m0 + mi * 16, k0 = n0 + ni * 16;
      const int r0 = k0 - q0 + (Pc - 15); // rel row base, in [1, 2017]
      v8f t0 = zf8(), t1 = zf8();
      #pragma unroll
      for (int kb = 0; kb < DHc; kb += 32) {
        v16bf aq = load_a(Q, Hc, q0 + ln, kb, half);
        int c1 = r0 + 16 + ln; if (c1 > 2 * Pc - 1) c1 = 2 * Pc - 1; // clamp edge
        v16bf br0 = load_b(relw, DHc, r0 + ln, kb, half);
        v16bf br1 = load_b(relw, DHc, c1, kb, half);
        t0 = wmma32(aq, br0, t0);
        t1 = wmma32(aq, br1, t1);
      }
      const int col = k0 + ln;
      const float mb = (1.0f - mask[b * Sc + col]) * -3.4028234663852886e38f;
      #pragma unroll
      for (int r = 0; r < 8; ++r) {
        const int i = r + 8 * half;
        const int cc = ln - i + 15;              // band column
        const int srcl = (cc & 15) + 16 * half;  // source lane of T[i][cc]
        float g0 = __shfl(t0[r], srcl, 32);
        float g1 = __shfl(t1[r], srcl, 32);
        float rv = (cc < 16) ? g0 : g1;
        const int row = q0 + i;
        sc[((size_t)bh * Sc + row) * Sc + col] = acc[mi][ni][r] * 0.125f + rv + mb;
      }
    }
  }
}

// ---------------------------------------------------------------------------
// Row softmax over S=1024, output bf16 probs. grid = B*NH*S blocks.
// ---------------------------------------------------------------------------
__global__ void k_softmax(const float* __restrict__ sc, __bf16* __restrict__ pr) {
  const size_t row = blockIdx.x;
  const int tid = threadIdx.x;
  const float4 v = ((const float4*)(sc + row * Sc))[tid];
  __shared__ float sm[8];
  float mx = fmaxf(fmaxf(v.x, v.y), fmaxf(v.z, v.w));
  mx = wave_max(mx);
  if ((tid & 31) == 0) sm[tid >> 5] = mx;
  __syncthreads();
  float M = sm[0];
  #pragma unroll
  for (int i = 1; i < 8; ++i) M = fmaxf(M, sm[i]);
  __syncthreads();
  float e0 = __expf(v.x - M), e1 = __expf(v.y - M);
  float e2 = __expf(v.z - M), e3 = __expf(v.w - M);
  float s = wave_sum(e0 + e1 + e2 + e3);
  if ((tid & 31) == 0) sm[tid >> 5] = s;
  __syncthreads();
  float T = 0.f;
  #pragma unroll
  for (int i = 0; i < 8; ++i) T += sm[i];
  const float inv = 1.0f / T;
  __bf16* o = pr + row * Sc + tid * 4;
  o[0] = (__bf16)(e0 * inv); o[1] = (__bf16)(e1 * inv);
  o[2] = (__bf16)(e2 * inv); o[3] = (__bf16)(e3 * inv);
}

// ---------------------------------------------------------------------------
// ctx = probs @ V : per (b,h), M=S, N=DH=64, K=S. V pre-transposed to (B,H,S).
// V tile (64 rows x 64 K) async-staged in LDS. grid = (S/128, 1, B*NH).
// ---------------------------------------------------------------------------
__global__ void k_ctx(const __bf16* __restrict__ pr, const __bf16* __restrict__ vt,
                      __bf16* __restrict__ ctxb) {
  __shared__ __bf16 bsm[2][64][64];
  const int bh = blockIdx.z, b = bh >> 4, h = bh & 15;
  const __bf16* A = pr + (size_t)bh * Sc * Sc;
  const __bf16* W = vt + ((size_t)b * Hc + h * DHc) * Sc;
  const int tid = threadIdx.x, lane = tid & 31, wv = tid >> 5;
  const int half = lane >> 4, ln = lane & 15;
  const int m0 = blockIdx.x * 128 + (wv & 3) * 32;
  const int n0 = (wv >> 2) * 32; // N = 64 covered by the two wave columns

  stage64(bsm, 0, W, Sc, 0, 0, tid);

  v8f acc[2][2];
  acc[0][0] = zf8(); acc[0][1] = zf8(); acc[1][0] = zf8(); acc[1][1] = zf8();
  int buf = 0;
  for (int kb = 0; kb < Sc; kb += 64) {
    wait_async0();
    __syncthreads();
    if (kb + 64 < Sc) stage64(bsm, buf ^ 1, W, Sc, 0, kb + 64, tid);
    const int bcol = n0 + ln;
    #pragma unroll
    for (int ks = 0; ks < 64; ks += 32) {
      v16bf a0 = load_a(A, Sc, m0 + ln, kb + ks, half);
      v16bf a1 = load_a(A, Sc, m0 + 16 + ln, kb + ks, half);
      v16bf b0 = load_b_lds(&bsm[buf][bcol][0], ks, half);
      v16bf b1 = load_b_lds(&bsm[buf][bcol + 16][0], ks, half);
      acc[0][0] = wmma32(a0, b0, acc[0][0]);
      acc[0][1] = wmma32(a0, b1, acc[0][1]);
      acc[1][0] = wmma32(a1, b0, acc[1][0]);
      acc[1][1] = wmma32(a1, b1, acc[1][1]);
    }
    __syncthreads();
    buf ^= 1;
  }
  #pragma unroll
  for (int mi = 0; mi < 2; ++mi)
    #pragma unroll
    for (int ni = 0; ni < 2; ++ni)
      #pragma unroll
      for (int r = 0; r < 8; ++r) {
        int row = m0 + mi * 16 + r + 8 * half;
        int col = n0 + ni * 16 + ln;
        ctxb[((size_t)b * Sc + row) * Hc + h * DHc + col] = (__bf16)acc[mi][ni][r];
      }
}

// ---------------------------------------------------------------------------
// Kernel-3 conv as 3 accumulated GEMM taps; loop flattened over (tap, kb) so
// the async LDS staging pipelines across tap boundaries. Wt layout (3, N, K).
// GELU=true: out bf16 = gelu(acc + bias); else: f32 residual += acc + bias.
// grid = (S/128, N/64, B).
// ---------------------------------------------------------------------------
template <bool GELU>
__global__ void k_conv(const __bf16* __restrict__ Afull, const __bf16* __restrict__ W3,
                       const float* __restrict__ bias, __bf16* __restrict__ outb,
                       float* __restrict__ outf, int K, int N) {
  __shared__ __bf16 bsm[2][64][64];
  const int bz = blockIdx.z;
  const __bf16* A = Afull + (size_t)bz * Sc * K;
  const int tid = threadIdx.x, lane = tid & 31, wv = tid >> 5;
  const int half = lane >> 4, ln = lane & 15;
  const int m0 = blockIdx.x * 128 + (wv & 3) * 32;
  const int nblk = blockIdx.y * 64;
  const int n0 = nblk + (wv >> 2) * 32;
  const int steps = 3 * (K >> 6);
  const v16bf za = zb16();

  stage64(bsm, 0, W3, K, nblk, 0, tid);

  v8f acc[2][2];
  acc[0][0] = zf8(); acc[0][1] = zf8(); acc[1][0] = zf8(); acc[1][1] = zf8();
  int buf = 0, tap = 0, kb = 0;
  for (int s = 0; s < steps; ++s) {
    wait_async0();
    __syncthreads();
    int ntap = tap, nkb = kb + 64;
    if (nkb == K) { nkb = 0; ++ntap; }
    if (s + 1 < steps) {
      stage64(bsm, buf ^ 1, W3 + (size_t)ntap * N * K, K, nblk, nkb, tid);
    }
    const int r0 = m0 + ln + tap - 1;
    const int r1 = r0 + 16;
    const bool ok0 = (unsigned)r0 < (unsigned)Sc;
    const bool ok1 = (unsigned)r1 < (unsigned)Sc;
    const int cr0 = ok0 ? r0 : 0;
    const int cr1 = ok1 ? r1 : 0;
    const int bcol = (wv >> 2) * 32 + ln;
    #pragma unroll
    for (int ks = 0; ks < 64; ks += 32) {
      v16bf a0 = load_a(A, K, cr0, kb + ks, half); if (!ok0) a0 = za;
      v16bf a1 = load_a(A, K, cr1, kb + ks, half); if (!ok1) a1 = za;
      v16bf b0 = load_b_lds(&bsm[buf][bcol][0], ks, half);
      v16bf b1 = load_b_lds(&bsm[buf][bcol + 16][0], ks, half);
      acc[0][0] = wmma32(a0, b0, acc[0][0]);
      acc[0][1] = wmma32(a0, b1, acc[0][1]);
      acc[1][0] = wmma32(a1, b0, acc[1][0]);
      acc[1][1] = wmma32(a1, b1, acc[1][1]);
    }
    __syncthreads();
    buf ^= 1; tap = ntap; kb = nkb;
  }
  #pragma unroll
  for (int mi = 0; mi < 2; ++mi)
    #pragma unroll
    for (int ni = 0; ni < 2; ++ni)
      #pragma unroll
      for (int r = 0; r < 8; ++r) {
        int row = m0 + mi * 16 + r + 8 * half;
        int col = n0 + ni * 16 + ln;
        float z = acc[mi][ni][r] + bias[col];
        size_t o = ((size_t)bz * Sc + row) * N + col;
        if (GELU) {
          outb[o] = (__bf16)(0.5f * z * (1.0f + erff(z * 0.70710678118654752f)));
        } else {
          outf[o] += z;
        }
      }
}

// ---------------------------------------------------------------------------
extern "C" void kernel_launch(void* const* d_in, const int* in_sizes, int n_in,
                              void* d_out, int out_size, void* d_ws, size_t ws_size,
                              hipStream_t stream) {
  (void)in_sizes; (void)n_in; (void)out_size; (void)ws_size;
  const int*   ids  = (const int*)d_in[0];
  const float* mask = (const float*)d_in[1];
  const float* emb  = (const float*)d_in[2];
  const float* Wq   = (const float*)d_in[3];
  const float* Wk   = (const float*)d_in[4];
  const float* Wv   = (const float*)d_in[5];
  const float* Wo   = (const float*)d_in[6];
  const float* bo   = (const float*)d_in[7];
  const float* rel  = (const float*)d_in[8];
  const float* c1w  = (const float*)d_in[9];
  const float* c1b  = (const float*)d_in[10];
  const float* c2w  = (const float*)d_in[11];
  const float* c2b  = (const float*)d_in[12];
  const float* l1s  = (const float*)d_in[13];
  const float* l1b  = (const float*)d_in[14];
  const float* l2s  = (const float*)d_in[15];
  const float* l2b  = (const float*)d_in[16];
  float* x = (float*)d_out; // (B,S,H) f32, residual stream, final output

  // ---- workspace carve-up ----
  char* w = (char*)d_ws;
  auto carve = [&](size_t bytes) {
    char* p = w; w += (bytes + 255) & ~(size_t)255; return p;
  };
  const size_t LHH  = (size_t)Lc * Hc * Hc;       // 4 Mi elems
  const size_t RELN = (size_t)Lc * 2 * Pc * DHc;  // 512 Ki
  const size_t C1N  = (size_t)Lc * Ic * Hc * 3;   // 48 Mi
  const size_t C2N  = (size_t)Lc * Hc * Ic * 3;   // 48 Mi
  const size_t TOK  = (size_t)Bc * Sc;            // 4096 tokens
  const size_t ACT  = TOK * Hc;                   // 4 Mi
  const size_t SCN  = (size_t)Bc * NHc * Sc * Sc; // 64 Mi

  __bf16* wqb   = (__bf16*)carve(LHH * 2);
  __bf16* wkb   = (__bf16*)carve(LHH * 2);
  __bf16* wvb   = (__bf16*)carve(LHH * 2);
  __bf16* wob   = (__bf16*)carve(LHH * 2);
  __bf16* relb  = (__bf16*)carve(RELN * 2);
  __bf16* c1wp  = (__bf16*)carve(C1N * 2);
  __bf16* c2wp  = (__bf16*)carve(C2N * 2);
  __bf16* hb    = (__bf16*)carve(ACT * 2);
  __bf16* qb    = (__bf16*)carve(ACT * 2);
  __bf16* kbuf  = (__bf16*)carve(ACT * 2);
  __bf16* vtb   = (__bf16*)carve(ACT * 2);
  __bf16* ctxb  = (__bf16*)carve(ACT * 2);
  float*  scf   = (float*)carve(SCN * 4);
  __bf16* prb   = (__bf16*)carve(SCN * 2);
  __bf16* y1b   = (__bf16*)carve(TOK * (size_t)Ic * 2);

  const dim3 blk(256);
  auto eb = [](long long n) { return dim3((unsigned)((n + 255) / 256)); };

  // ---- one-time precision conversion / repacking ----
  k_cast<<<eb(LHH), blk, 0, stream>>>(Wq, wqb, (long long)LHH);
  k_cast<<<eb(LHH), blk, 0, stream>>>(Wk, wkb, (long long)LHH);
  k_cast<<<eb(LHH), blk, 0, stream>>>(Wv, wvb, (long long)LHH);
  k_cast<<<eb(LHH), blk, 0, stream>>>(Wo, wob, (long long)LHH);
  k_cast<<<eb(RELN), blk, 0, stream>>>(rel, relb, (long long)RELN);
  k_pack3<<<eb(C1N), blk, 0, stream>>>(c1w, c1wp, Ic, Hc, (long long)C1N);
  k_pack3<<<eb(C2N), blk, 0, stream>>>(c2w, c2wp, Hc, Ic, (long long)C2N);

  // ---- embedding ----
  k_embed<<<eb((long long)ACT), blk, 0, stream>>>(ids, emb, x);

  // ---- layers ----
  for (int l = 0; l < Lc; ++l) {
    const __bf16* wq_l = wqb + (size_t)l * Hc * Hc;
    const __bf16* wk_l = wkb + (size_t)l * Hc * Hc;
    const __bf16* wv_l = wvb + (size_t)l * Hc * Hc;
    const __bf16* wo_l = wob + (size_t)l * Hc * Hc;
    const __bf16* rel_l = relb + (size_t)l * 2 * Pc * DHc;
    const __bf16* c1_l = c1wp + (size_t)l * 3 * Ic * Hc;
    const __bf16* c2_l = c2wp + (size_t)l * 3 * Hc * Ic;

    // LN1 -> h (bf16)
    k_layernorm<<<dim3((unsigned)TOK), blk, 0, stream>>>(x, l1s + l * Hc, l1b + l * Hc, hb);

    // Q, K (row-major bf16), V (transposed to (B,H,S))
    k_gemm<0><<<dim3(32, 16), blk, 0, stream>>>(hb, wq_l, nullptr, qb, nullptr, Hc, Hc);
    k_gemm<0><<<dim3(32, 16), blk, 0, stream>>>(hb, wk_l, nullptr, kbuf, nullptr, Hc, Hc);
    k_gemm<1><<<dim3(32, 16), blk, 0, stream>>>(hb, wv_l, nullptr, vtb, nullptr, Hc, Hc);

    // scores = QK^T/8 + rel + mask ; softmax -> probs bf16 ; ctx = P@V
    k_scores<<<dim3(8, 16, Bc * NHc), blk, 0, stream>>>(qb, kbuf, rel_l, mask, scf);
    k_softmax<<<dim3((unsigned)(Bc * NHc * Sc)), blk, 0, stream>>>(scf, prb);
    k_ctx<<<dim3(8, 1, Bc * NHc), blk, 0, stream>>>(prb, vtb, ctxb);

    // attention output projection + residual into x
    k_gemm<2><<<dim3(32, 16), blk, 0, stream>>>(ctxb, wo_l, bo + l * Hc, nullptr, x, Hc, Hc);

    // LN2 -> h2 (reuse hb)
    k_layernorm<<<dim3((unsigned)TOK), blk, 0, stream>>>(x, l2s + l * Hc, l2b + l * Hc, hb);

    // conv1 (H->I, k=3) + exact GELU -> y1 bf16 ; conv2 (I->H, k=3) + residual
    k_conv<true><<<dim3(8, 64, Bc), blk, 0, stream>>>(hb, c1_l, c1b + l * Ic, y1b, nullptr, Hc, Ic);
    k_conv<false><<<dim3(8, 16, Bc), blk, 0, stream>>>(y1b, c2_l, c2b + l * Hc, nullptr, x, Ic, Hc);
  }
}
